// CAF_41463614275926
// MI455X (gfx1250) — matrix operand order repaired and checked by reading, not compiled, not run on hardware
//
#include <hip/hip_runtime.h>
#include <hip/hip_bf16.h>

typedef __attribute__((ext_vector_type(16))) _Float16 v16h;
typedef __attribute__((ext_vector_type(8)))  float    v8f;

#define HWSZ 4096
#define LANE (threadIdx.x & 31)

// ---------------------------------------------------------------------------
// Branchless WMMA operand loaders, k = 16*(lane/16) + element (f16 16x16x32, wave32).
// ---------------------------------------------------------------------------

// A[m,k] = W[(m0+m)*ldw + k0+k]; requires 16B alignment of W + row*ldw + k0.
__device__ __forceinline__ v16h load_amat(const float* __restrict__ W, int ldw,
                                          int m0, int k0) {
  const int l = LANE;
  const int row = m0 + (l & 15);
  const int koff = (l & 16) ? 16 : 0;
  const float* p = W + (size_t)row * ldw + k0 + koff;
  const float4 x0 = ((const float4*)p)[0];
  const float4 x1 = ((const float4*)p)[1];
  const float4 x2 = ((const float4*)p)[2];
  const float4 x3 = ((const float4*)p)[3];
  v16h a;
  a[0] = (_Float16)x0.x;  a[1] = (_Float16)x0.y;  a[2] = (_Float16)x0.z;  a[3] = (_Float16)x0.w;
  a[4] = (_Float16)x1.x;  a[5] = (_Float16)x1.y;  a[6] = (_Float16)x1.z;  a[7] = (_Float16)x1.w;
  a[8] = (_Float16)x2.x;  a[9] = (_Float16)x2.y;  a[10] = (_Float16)x2.z; a[11] = (_Float16)x2.w;
  a[12] = (_Float16)x3.x; a[13] = (_Float16)x3.y; a[14] = (_Float16)x3.z; a[15] = (_Float16)x3.w;
  return a;
}

// B[k,n] = X[(k0+k)*ldx + n0+n]  (row-strided, coalesced across the 16-lane group)
__device__ __forceinline__ v16h load_bmat(const float* __restrict__ X, int ldx,
                                          int k0, int n0) {
  const int l = LANE;
  const float* p = X + (size_t)(k0 + ((l & 16) ? 16 : 0)) * ldx + n0 + (l & 15);
  v16h b;
#pragma unroll
  for (int v = 0; v < 8; ++v) {
    b[2 * v]     = (_Float16)p[(size_t)(2 * v) * ldx];
    b[2 * v + 1] = (_Float16)p[(size_t)(2 * v + 1) * ldx];
  }
  return b;
}

// B operand from a transposed 32-col matrix: B[k,n] = XT[(m0+n)*32 + k]
__device__ __forceinline__ v16h load_bT32(const float* __restrict__ XT, int m0) {
  const int l = LANE;
  const float* p = XT + (size_t)(m0 + (l & 15)) * 32 + ((l & 16) ? 16 : 0);
  const float4 x0 = ((const float4*)p)[0];
  const float4 x1 = ((const float4*)p)[1];
  const float4 x2 = ((const float4*)p)[2];
  const float4 x3 = ((const float4*)p)[3];
  v16h b;
  b[0] = (_Float16)x0.x;  b[1] = (_Float16)x0.y;  b[2] = (_Float16)x0.z;  b[3] = (_Float16)x0.w;
  b[4] = (_Float16)x1.x;  b[5] = (_Float16)x1.y;  b[6] = (_Float16)x1.z;  b[7] = (_Float16)x1.w;
  b[8] = (_Float16)x2.x;  b[9] = (_Float16)x2.y;  b[10] = (_Float16)x2.z; b[11] = (_Float16)x2.w;
  b[12] = (_Float16)x3.x; b[13] = (_Float16)x3.y; b[14] = (_Float16)x3.z; b[15] = (_Float16)x3.w;
  return b;
}

// ---------------------------------------------------------------------------
// Async global->LDS staging (CDNA5): GLOBAL_LOAD_ASYNC_TO_LDS_B128, ASYNCcnt.
// LDS dest address = low 32 bits of the flat shared-memory pointer (ISA 10.2).
// ---------------------------------------------------------------------------
__device__ __forceinline__ void async_g2l_b128(const float* g, const float* l) {
  const unsigned long long ga = (unsigned long long)(uintptr_t)g;
  const unsigned int la = (unsigned int)(uintptr_t)l;
  asm volatile("global_load_async_to_lds_b128 %0, %1, off" ::"v"(la), "v"(ga)
               : "memory");
}
__device__ __forceinline__ void wait_async_le16() {
  asm volatile("s_wait_asynccnt 0x10" ::: "memory");
}
__device__ __forceinline__ void wait_async_0() {
  asm volatile("s_wait_asynccnt 0x0" ::: "memory");
}

// copy V[0:64][m0:m0+32] (row stride HWSZ) into lds tile [64][32]; 16 b128/lane-set
__device__ __forceinline__ void stage_v_tile(const float* __restrict__ Vm0,
                                             float* __restrict__ lds) {
  const int l = LANE;
#pragma unroll
  for (int i = 0; i < 16; ++i) {
    const int ch = i * 32 + l;        // 512 16B-chunks total
    const int c = ch >> 3;            // row 0..63
    const int sub = (ch & 7) * 4;     // float offset within row
    async_g2l_b128(Vm0 + (size_t)c * HWSZ + sub, lds + c * 32 + sub);
  }
}

// ---------------------------------------------------------------------------
// Generic per-batch GEMM: D[z] = W[M,K]*X[z][K,N] + bias, M = 16*MT per wave.
// ---------------------------------------------------------------------------
template <int MT>
__global__ void gemm_wmma(const float* __restrict__ X, const float* __restrict__ W,
                          const float* __restrict__ bias, float* __restrict__ D,
                          int K, int N, long xstride, long dstride) {
  const int n0 = blockIdx.x * 16;
  const int m0 = blockIdx.y * (16 * MT);
  const float* Xb = X + (size_t)blockIdx.z * xstride;
  float* Db = D + (size_t)blockIdx.z * dstride;

  v8f acc[MT];
#pragma unroll
  for (int t = 0; t < MT; ++t) acc[t] = (v8f){0.f, 0.f, 0.f, 0.f, 0.f, 0.f, 0.f, 0.f};

  for (int k0 = 0; k0 < K; k0 += 32) {
    const v16h b = load_bmat(Xb, N, k0, n0);
#pragma unroll
    for (int t = 0; t < MT; ++t) {
      const v16h a = load_amat(W, K, m0 + t * 16, k0);
      acc[t] = __builtin_amdgcn_wmma_f32_16x16x32_f16(false, a, false, b, (short)0,
                                                      acc[t], false, false);
    }
  }
  const int l = LANE;
  const int col = n0 + (l & 15);
  const int rb = (l & 16) ? 8 : 0;
#pragma unroll
  for (int t = 0; t < MT; ++t)
#pragma unroll
    for (int r = 0; r < 8; ++r) {
      const int row = m0 + t * 16 + rb + r;
      Db[(size_t)row * N + col] = acc[t][r] + bias[row];
    }
}

// q/k GEMM (16 padded rows) storing TRANSPOSED into DT[HW][32] (rows 16..31
// pre-zeroed). grid (N/16, 1, B), block 32.
__global__ void gemm_qk_t(const float* __restrict__ X, const float* __restrict__ W,
                          const float* __restrict__ bias, float* __restrict__ DT,
                          int K, int N, long xstride, long dtstride) {
  const int n0 = blockIdx.x * 16;
  const float* Xb = X + (size_t)blockIdx.z * xstride;
  float* Dt = DT + (size_t)blockIdx.z * dtstride;

  v8f acc = {0.f, 0.f, 0.f, 0.f, 0.f, 0.f, 0.f, 0.f};
  for (int k0 = 0; k0 < K; k0 += 32) {
    const v16h a = load_amat(W, K, 0, k0);
    const v16h b = load_bmat(Xb, N, k0, n0);
    acc = __builtin_amdgcn_wmma_f32_16x16x32_f16(false, a, false, b, (short)0, acc,
                                                 false, false);
  }
  const int l = LANE;
  const int col = n0 + (l & 15);
  const int rb = (l & 16) ? 8 : 0;
  float4 o0 = {acc[0] + bias[rb + 0], acc[1] + bias[rb + 1],
               acc[2] + bias[rb + 2], acc[3] + bias[rb + 3]};
  float4 o1 = {acc[4] + bias[rb + 4], acc[5] + bias[rb + 5],
               acc[6] + bias[rb + 6], acc[7] + bias[rb + 7]};
  float4* dst = (float4*)(Dt + (size_t)col * 32 + rb);
  dst[0] = o0;
  dst[1] = o1;
}

// ---------------------------------------------------------------------------
// small helpers
// ---------------------------------------------------------------------------
__global__ void zero_buf(float* __restrict__ p, int n) {
  const int i = blockIdx.x * 256 + threadIdx.x;
  if (i < n) p[i] = 0.f;
}

__global__ void pad_qk(const float* __restrict__ qw, const float* __restrict__ qb,
                       const float* __restrict__ kw, const float* __restrict__ kb,
                       float* __restrict__ WQ, float* __restrict__ BQ,
                       float* __restrict__ WK, float* __restrict__ BK) {
  const int i = blockIdx.x * 256 + threadIdx.x;  // 0..1023
  if (i < 1024) {
    const bool v = i < 512;
    WQ[i] = v ? qw[i] : 0.f;
    WK[i] = v ? kw[i] : 0.f;
  }
  if (i < 16) {
    BQ[i] = (i < 8) ? qb[i] : 0.f;
    BK[i] = (i < 8) ? kb[i] : 0.f;
  }
}

// ---------------------------------------------------------------------------
// Channel attention
// ---------------------------------------------------------------------------
__global__ void chan_avg(const float* __restrict__ P, float* __restrict__ avg) {
  __shared__ float red[256];
  const int ch = blockIdx.x;  // 0..511
  const float* p = P + (size_t)ch * HWSZ;
  float s = 0.f;
  for (int i = threadIdx.x; i < HWSZ; i += 256) s += p[i];
  red[threadIdx.x] = s;
  __syncthreads();
  for (int st = 128; st > 0; st >>= 1) {
    if ((int)threadIdx.x < st) red[threadIdx.x] += red[threadIdx.x + st];
    __syncthreads();
  }
  if (threadIdx.x == 0) avg[ch] = red[0] * (1.f / (float)HWSZ);
}

__global__ void ca_mlp(const float* __restrict__ avg, const float* __restrict__ w1,
                       const float* __restrict__ w2, float* __restrict__ cas) {
  const int ib = blockIdx.x;
  const float* a = avg + ib * 64;
  const int c = threadIdx.x;
  float h[4];
#pragma unroll
  for (int r = 0; r < 4; ++r) {
    float t = 0.f;
    for (int cc = 0; cc < 64; ++cc) t += a[cc] * w1[r * 64 + cc];
    h[r] = fmaxf(t, 0.f);
  }
  float t = 0.f;
#pragma unroll
  for (int r = 0; r < 4; ++r) t += h[r] * w2[c * 4 + r];
  cas[ib * 64 + c] = 1.f / (1.f + __expf(-t));
}

// ---------------------------------------------------------------------------
// Spatial attention
// ---------------------------------------------------------------------------
__global__ void sa_pool(const float* __restrict__ P, float* __restrict__ am) {
  const int idx = blockIdx.x * 256 + threadIdx.x;
  const int ib = idx >> 12;
  const int hw = idx & 4095;
  const float* p = P + (size_t)ib * 64 * HWSZ + hw;
  float s = 0.f, mx = -1e30f;
  for (int c = 0; c < 64; ++c) {
    const float v = p[(size_t)c * HWSZ];
    s += v;
    mx = fmaxf(mx, v);
  }
  am[(size_t)ib * 2 * HWSZ + hw] = s * (1.f / 64.f);
  am[(size_t)ib * 2 * HWSZ + HWSZ + hw] = mx;
}

__global__ void sa_conv(const float* __restrict__ am, const float* __restrict__ w,
                        const float* __restrict__ bb, float* __restrict__ sa) {
  const int idx = blockIdx.x * 256 + threadIdx.x;
  const int ib = idx >> 12;
  const int hw = idx & 4095;
  const int y = hw >> 6, x = hw & 63;
  const float* a0 = am + (size_t)ib * 2 * HWSZ;
  float acc = bb[0];
  for (int ch = 0; ch < 2; ++ch)
    for (int ky = 0; ky < 7; ++ky) {
      const int yy = y + ky - 3;
      if (yy < 0 || yy > 63) continue;
      for (int kx = 0; kx < 7; ++kx) {
        const int xx = x + kx - 3;
        if (xx < 0 || xx > 63) continue;
        acc += a0[ch * HWSZ + yy * 64 + xx] * w[ch * 49 + ky * 7 + kx];
      }
    }
  sa[(size_t)ib * HWSZ + hw] = 1.f / (1.f + __expf(-acc));
}

__global__ void apply_gate(const float* __restrict__ P, const float* __restrict__ cas,
                           const float* __restrict__ sa, float* __restrict__ ES,
                           float* __restrict__ CAT) {
  const size_t idx = (size_t)blockIdx.x * 256 + threadIdx.x;
  const int hw = (int)(idx & 4095);
  const int c = (int)((idx >> 12) & 63);
  const int ib = (int)(idx >> 18);  // 0-3 swin, 4-7 resnet
  const float v = P[idx] * cas[ib * 64 + c] * sa[(size_t)ib * HWSZ + hw];
  ES[idx] = v;
  if (ib < 4) CAT[(size_t)ib * 128 * HWSZ + (size_t)(64 + c) * HWSZ + hw] = v;
}

// ---------------------------------------------------------------------------
// Fused cross-attention. QT/KT are [B][HW][32] transposed, zero-padded c>=8.
// V tiles double-buffered in LDS via async global->LDS DMA (ASYNCcnt).
// grid (HW/16, B), block 32 (one wave). Writes gamma*out + es into CAT[:,0:64].
// ---------------------------------------------------------------------------
__global__ void attn_fused(const float* __restrict__ QT, const float* __restrict__ KT,
                           const float* __restrict__ V, const float* __restrict__ ES,
                           const float* __restrict__ gamma, float* __restrict__ CAT) {
  const int n0 = blockIdx.x * 16;
  const int b = blockIdx.y;
  const int l = LANE;
  const float* QTb = QT + (size_t)b * HWSZ * 32;
  const float* KTb = KT + (size_t)b * HWSZ * 32;
  const float* Vb = V + (size_t)b * 64 * HWSZ;
  const float* Eb = ES + (size_t)b * 64 * HWSZ;
  float* Cb = CAT + (size_t)b * 128 * HWSZ;

  __shared__ float rm[16];
  __shared__ float rs[16];
  __shared__ _Float16 pb[16 * 32];
  __shared__ float ldsV[2][64 * 32];  // double-buffered V tile [c][m]

  const v16h aQ = load_amat(QTb, 32, n0, 0);
  const v8f zero = {0.f, 0.f, 0.f, 0.f, 0.f, 0.f, 0.f, 0.f};

  // ---- pass 1: row max ----
  float rmax[8];
#pragma unroll
  for (int r = 0; r < 8; ++r) rmax[r] = -1e30f;
  for (int m0 = 0; m0 < HWSZ; m0 += 16) {
    const v16h bK = load_bT32(KTb, m0);
    const v8f s = __builtin_amdgcn_wmma_f32_16x16x32_f16(false, aQ, false, bK, (short)0,
                                                         zero, false, false);
#pragma unroll
    for (int r = 0; r < 8; ++r) rmax[r] = fmaxf(rmax[r], s[r]);
  }
#pragma unroll
  for (int r = 0; r < 8; ++r)
    for (int m = 8; m >= 1; m >>= 1) rmax[r] = fmaxf(rmax[r], __shfl_xor(rmax[r], m, 32));
  if ((l & 15) == 0) {
    const int off = (l & 16) ? 8 : 0;
#pragma unroll
    for (int r = 0; r < 8; ++r) rm[off + r] = rmax[r];
  }
  __syncthreads();
  float rmr[8];
#pragma unroll
  for (int r = 0; r < 8; ++r) rmr[r] = rm[r + ((l & 16) ? 8 : 0)];

  // ---- pass 2: unnormalized P, O += V * P^T, V staged via async DMA ----
  v8f oacc[4];
#pragma unroll
  for (int t = 0; t < 4; ++t) oacc[t] = zero;
  float rsum[8];
#pragma unroll
  for (int r = 0; r < 8; ++r) rsum[r] = 0.f;

  stage_v_tile(Vb, ldsV[0]);  // prologue: block 0

  for (int m0 = 0; m0 < HWSZ; m0 += 32) {
    const int cur = (m0 >> 5) & 1;
    const bool more = (m0 + 32) < HWSZ;
    if (more) stage_v_tile(Vb + m0 + 32, ldsV[cur ^ 1]);  // prefetch next tile

#pragma unroll
    for (int t = 0; t < 2; ++t) {
      const v16h bK = load_bT32(KTb, m0 + t * 16);
      const v8f s = __builtin_amdgcn_wmma_f32_16x16x32_f16(false, aQ, false, bK,
                                                           (short)0, zero, false, false);
#pragma unroll
      for (int r = 0; r < 8; ++r) {
        const float p = __expf(s[r] - rmr[r]);
        rsum[r] += p;
        const int n = r + ((l & 16) ? 8 : 0);
        const int mm = t * 16 + (l & 15);
        pb[n * 32 + mm] = (_Float16)p;
      }
    }
    // current tile's 16 async b128 are the oldest outstanding ops (in-order):
    if (more) wait_async_le16();
    else wait_async_0();
    __syncthreads();
    v16h bP;
    {
      const int coln = l & 15;
      const int mb = (l & 16) ? 16 : 0;
#pragma unroll
      for (int v = 0; v < 8; ++v) {
        bP[2 * v]     = pb[coln * 32 + mb + 2 * v];
        bP[2 * v + 1] = pb[coln * 32 + mb + 2 * v + 1];
      }
    }
#pragma unroll
    for (int ct = 0; ct < 4; ++ct) {
      const v16h aV = load_amat(ldsV[cur], 32, ct * 16, 0);  // ds_load path
      oacc[ct] = __builtin_amdgcn_wmma_f32_16x16x32_f16(false, aV, false, bP, (short)0,
                                                        oacc[ct], false, false);
    }
    __syncthreads();
  }

  // ---- normalize + fuse gamma*out + es ----
#pragma unroll
  for (int r = 0; r < 8; ++r)
    for (int m = 8; m >= 1; m >>= 1) rsum[r] += __shfl_xor(rsum[r], m, 32);
  if ((l & 15) == 0) {
    const int off = (l & 16) ? 8 : 0;
#pragma unroll
    for (int r = 0; r < 8; ++r) rs[off + r] = rsum[r];
  }
  __syncthreads();
  const float inv = 1.f / rs[l & 15];
  const float g = gamma[0];
  const int n = n0 + (l & 15);
#pragma unroll
  for (int ct = 0; ct < 4; ++ct)
#pragma unroll
    for (int r = 0; r < 8; ++r) {
      const int c = ct * 16 + r + ((l & 16) ? 8 : 0);
      Cb[(size_t)c * HWSZ + n] = g * (oacc[ct][r] * inv) + Eb[(size_t)c * HWSZ + n];
    }
}

// ---------------------------------------------------------------------------
// BatchNorm (batch stats) + ReLU
// ---------------------------------------------------------------------------
__global__ void bn_stats(const float* __restrict__ Y, float* __restrict__ stats) {
  __shared__ float rsh[256], rqh[256];
  const int c = blockIdx.x;  // 0..127
  float s = 0.f, q = 0.f;
  for (int i = threadIdx.x; i < 4 * HWSZ; i += 256) {
    const int b = i >> 12;
    const int hw = i & 4095;
    const float v = Y[(size_t)b * 128 * HWSZ + (size_t)c * HWSZ + hw];
    s += v;
    q += v * v;
  }
  rsh[threadIdx.x] = s;
  rqh[threadIdx.x] = q;
  __syncthreads();
  for (int st = 128; st > 0; st >>= 1) {
    if ((int)threadIdx.x < st) {
      rsh[threadIdx.x] += rsh[threadIdx.x + st];
      rqh[threadIdx.x] += rqh[threadIdx.x + st];
    }
    __syncthreads();
  }
  if (threadIdx.x == 0) {
    const float mean = rsh[0] * (1.f / 16384.f);
    const float var = rqh[0] * (1.f / 16384.f) - mean * mean;
    stats[c] = mean;
    stats[128 + c] = rsqrtf(var + 1e-5f);
  }
}

__global__ void bn_apply(const float* __restrict__ Y, const float* __restrict__ stats,
                         const float* __restrict__ sc, const float* __restrict__ bi,
                         float* __restrict__ out) {
  const size_t idx = (size_t)blockIdx.x * 256 + threadIdx.x;
  const int c = (int)((idx >> 12) & 127);
  const float v = (Y[idx] - stats[c]) * stats[128 + c] * sc[c] + bi[c];
  out[idx] = fmaxf(v, 0.f);
}

// ---------------------------------------------------------------------------
extern "C" void kernel_launch(void* const* d_in, const int* in_sizes, int n_in,
                              void* d_out, int out_size, void* d_ws, size_t ws_size,
                              hipStream_t stream) {
  const float* swin   = (const float*)d_in[0];
  const float* resnet = (const float*)d_in[1];
  const float* proj_w = (const float*)d_in[2];
  const float* proj_b = (const float*)d_in[3];
  const float* ca_w1  = (const float*)d_in[4];
  const float* ca_w2  = (const float*)d_in[5];
  const float* sa_w   = (const float*)d_in[6];
  const float* sa_b   = (const float*)d_in[7];
  const float* q_w    = (const float*)d_in[8];
  const float* q_b    = (const float*)d_in[9];
  const float* k_w    = (const float*)d_in[10];
  const float* k_b    = (const float*)d_in[11];
  const float* v_w    = (const float*)d_in[12];
  const float* v_b    = (const float*)d_in[13];
  const float* gamma  = (const float*)d_in[14];
  const float* ref_w  = (const float*)d_in[15];
  const float* ref_b  = (const float*)d_in[16];
  const float* bn_sc  = (const float*)d_in[17];
  const float* bn_bi  = (const float*)d_in[18];

  float* ws = (float*)d_ws;
  const size_t nPB = (size_t)2 * 4 * 64 * HWSZ;    // 2,097,152
  const size_t nQT = (size_t)4 * HWSZ * 32;        // 524,288
  float* P    = ws;                                // [2][B][64][HW]
  float* ES   = P + nPB;                           // [2][B][64][HW]
  float* CAT  = ES + nPB;                          // [B][128][HW]
  float* Y    = CAT + nPB;                         // [B][128][HW]
  float* QT   = Y + nPB;                           // [B][HW][32] padded
  float* KT   = QT + nQT;                          // [B][HW][32] padded
  float* Vb   = KT + nQT;                          // [B][64][HW]
  float* AVG  = Vb + (size_t)4 * 64 * HWSZ;        // [512]
  float* CAS  = AVG + 512;
  float* AM   = CAS + 512;                         // [2B][2][HW]
  float* SA   = AM + (size_t)8 * 2 * HWSZ;         // [2B][HW]
  float* STAT = SA + (size_t)8 * HWSZ;             // [256]
  float* WQP  = STAT + 256;                        // [16][64]
  float* WKP  = WQP + 1024;                        // [16][64]
  float* BQP  = WKP + 1024;                        // [16]
  float* BKP  = BQP + 16;                          // [16]

  const float* ER = ES + (size_t)4 * 64 * HWSZ;

  // 0) pad q/k weights; zero QT/KT (contiguous)
  pad_qk<<<4, 256, 0, stream>>>(q_w, q_b, k_w, k_b, WQP, BQP, WKP, BKP);
  zero_buf<<<(int)(2 * nQT / 256), 256, 0, stream>>>(QT, (int)(2 * nQT));

  // 1) projections
  gemm_wmma<4><<<dim3(256, 1, 4), 32, 0, stream>>>(swin, proj_w, proj_b, P, 128, HWSZ,
                                                   128L * HWSZ, 64L * HWSZ);
  gemm_wmma<4><<<dim3(256, 1, 4), 32, 0, stream>>>(resnet, proj_w, proj_b,
                                                   P + (size_t)4 * 64 * HWSZ, 128, HWSZ,
                                                   128L * HWSZ, 64L * HWSZ);
  // 2) channel attention
  chan_avg<<<512, 256, 0, stream>>>(P, AVG);
  ca_mlp<<<8, 64, 0, stream>>>(AVG, ca_w1, ca_w2, CAS);
  // 3) spatial attention
  sa_pool<<<(8 * HWSZ) / 256, 256, 0, stream>>>(P, AM);
  sa_conv<<<(8 * HWSZ) / 256, 256, 0, stream>>>(AM, sa_w, sa_b, SA);
  // 4) gating (also fills CAT[:,64:128])
  apply_gate<<<(int)(nPB / 256), 256, 0, stream>>>(P, CAS, SA, ES, CAT);
  // 5) q from es, k/v from er  (q/k transposed+padded)
  gemm_qk_t<<<dim3(256, 1, 4), 32, 0, stream>>>(ES, WQP, BQP, QT, 64, HWSZ,
                                                64L * HWSZ, (long)HWSZ * 32);
  gemm_qk_t<<<dim3(256, 1, 4), 32, 0, stream>>>(ER, WKP, BKP, KT, 64, HWSZ,
                                                64L * HWSZ, (long)HWSZ * 32);
  gemm_wmma<4><<<dim3(256, 1, 4), 32, 0, stream>>>(ER, v_w, v_b, Vb, 64, HWSZ,
                                                   64L * HWSZ, 64L * HWSZ);
  // 6) fused cross-attention -> CAT[:,0:64]
  attn_fused<<<dim3(256, 4), 32, 0, stream>>>(QT, KT, Vb, ES, gamma, CAT);
  // 7) refine conv1x1
  gemm_wmma<4><<<dim3(256, 2, 4), 32, 0, stream>>>(CAT, ref_w, ref_b, Y, 128, HWSZ,
                                                   128L * HWSZ, 128L * HWSZ);
  // 8) BN + ReLU
  bn_stats<<<128, 256, 0, stream>>>(Y, STAT);
  bn_apply<<<(int)(nPB / 256), 256, 0, stream>>>(Y, STAT, bn_sc, bn_bi, (float*)d_out);
}
